// RNNCaptioning_91010357002762
// MI455X (gfx1250) — compile-verified, matrix operand order checked by reading, not run on hardware
//
#include <hip/hip_runtime.h>
#include <hip/hip_bf16.h>
#include <math.h>

// Problem sizes (fixed by the reference)
#define B_SZ   256
#define FT_SZ  2048
#define EMB_SZ 1024
#define MEM_SZ 1024
#define VOC_SZ 10000
#define T_SZ   20

typedef __bf16 bf16;
typedef __attribute__((ext_vector_type(8)))  __bf16 v8bf;
typedef __attribute__((ext_vector_type(16))) __bf16 v16bf;
typedef __attribute__((ext_vector_type(8)))  float  v8f;
typedef int v4i_ __attribute__((vector_size(16)));   // matches builtin param type

// GEMM tiling: block = 256 threads = 8 waves (wave32).
// Block tile 128(M) x 64(N), K-step 32. Wave grid 4(M) x 2(N),
// each wave owns 32x32 = 2x2 WMMA 16x16 fragments.
// Double-buffered LDS: stage tile k+1 while WMMAs consume tile k.
#define BM   128
#define BN   64
#define BK   32
#define LDSS 40                 // LDS row stride in bf16 (80B: 16B aligned, bank skew)
#define ASTRIDE (BM * LDSS)     // one A buffer, elements
#define BSTRIDE (BN * LDSS)     // one B buffer, elements

// ---- 16B global->LDS copy: CDNA5 async-to-LDS when available, sync fallback
#if __has_builtin(__builtin_amdgcn_global_load_async_to_lds_b128) && \
    __has_builtin(__builtin_amdgcn_s_wait_asynccnt)
#define USE_ASYNC_LDS 1
#else
#define USE_ASYNC_LDS 0
#endif

__device__ __forceinline__ void copy16_g2l(const bf16* g, bf16* l)
{
#if USE_ASYNC_LDS
  __builtin_amdgcn_global_load_async_to_lds_b128(
      (__attribute__((address_space(1))) v4i_*)g,
      (__attribute__((address_space(3))) v4i_*)l, 0, 0);
#else
  *(v8bf*)l = *(const v8bf*)g;
#endif
}

__device__ __forceinline__ void wait_async_lds()
{
#if USE_ASYNC_LDS
  __builtin_amdgcn_s_wait_asynccnt(0);
#endif
}

// ---------------------------------------------------------------------------
// Shared WMMA K-loop: acc += A[rowBase:+128, :K] * Bw[colBase:+64, :K]^T
// A is [M,K] bf16 row-major; Bw is [N,K] bf16 row-major (weight layout), so
// the B operand of WMMA (KxN) reads weight rows directly (K contiguous).
// NGUARD: clamp B rows to Nlimit-1 (vocab edge tile; extra cols never stored,
// and each output column depends only on its own B row).
// ---------------------------------------------------------------------------
template<bool NGUARD>
__device__ __forceinline__ void gemm_k_loop(
    const bf16* __restrict__ A, const bf16* __restrict__ Bw,
    int K, int ldA, int ldB, int Nlimit,
    int rowBase, int colBase,
    bf16* As, bf16* Bs, v8f (&acc)[2][2])
{
  const int tid  = threadIdx.x;
  const int lane = tid & 31;
  const int wave = tid >> 5;
  const int wm   = wave >> 1;      // 0..3 -> M sub-tile
  const int wn   = wave & 1;       // 0..1 -> N sub-tile
  const int half = lane >> 4;      // K-half selector per documented A layout
  const int l16  = lane & 15;      // row (A) / col (B) within fragment

  // Per-thread staging coordinates (16B chunks).
  const int ar0 = (tid      ) >> 2, akc0 = ((tid      ) & 3) * 8;
  const int ar1 = (tid + 256) >> 2, akc1 = ((tid + 256) & 3) * 8;
  int brow = colBase + (tid >> 2);
  if (NGUARD) brow = brow < Nlimit ? brow : (Nlimit - 1);
  const int bkc = (tid & 3) * 8;

  auto stage = [&](int k0, int buf) {
    copy16_g2l(A + (size_t)(rowBase + ar0) * ldA + k0 + akc0,
               As + buf * ASTRIDE + ar0 * LDSS + akc0);
    copy16_g2l(A + (size_t)(rowBase + ar1) * ldA + k0 + akc1,
               As + buf * ASTRIDE + ar1 * LDSS + akc1);
    copy16_g2l(Bw + (size_t)brow * ldB + k0 + bkc,
               Bs + buf * BSTRIDE + (tid >> 2) * LDSS + bkc);
  };

  stage(0, 0);
  wait_async_lds();
  __syncthreads();

  const int nsteps = K / BK;
  for (int s = 0; s < nsteps; ++s) {
    const int cur = s & 1;
    if (s + 1 < nsteps) stage((s + 1) * BK, cur ^ 1);   // prefetch next tile

    // ---- build fragments per ISA 16-bit A layout and issue 4 WMMAs
    const bf16* Ac = As + cur * ASTRIDE;
    const bf16* Bc = Bs + cur * BSTRIDE;
    v16bf af[2], bfr[2];
    #pragma unroll
    for (int i = 0; i < 2; ++i) {
      int ar = wm * 32 + i * 16 + l16;
      v8bf lo = *(const v8bf*)&Ac[ar * LDSS + half * 8];        // K 0..7 / 8..15
      v8bf hi = *(const v8bf*)&Ac[ar * LDSS + 16 + half * 8];   // K 16..23 / 24..31
      #pragma unroll
      for (int e = 0; e < 8; ++e) { af[i][e] = lo[e]; af[i][8 + e] = hi[e]; }
    }
    #pragma unroll
    for (int j = 0; j < 2; ++j) {
      int br = wn * 32 + j * 16 + l16;
      v8bf lo = *(const v8bf*)&Bc[br * LDSS + half * 8];
      v8bf hi = *(const v8bf*)&Bc[br * LDSS + 16 + half * 8];
      #pragma unroll
      for (int e = 0; e < 8; ++e) { bfr[j][e] = lo[e]; bfr[j][8 + e] = hi[e]; }
    }
    #pragma unroll
    for (int i = 0; i < 2; ++i)
      #pragma unroll
      for (int j = 0; j < 2; ++j)
        acc[i][j] = __builtin_amdgcn_wmma_f32_16x16x32_bf16(
            false, af[i], false, bfr[j], (short)0, acc[i][j], false, false);

    wait_async_lds();   // next tile fully landed in LDS
    __syncthreads();    // all waves done reading `cur` before it is re-staged
  }
}

__device__ __forceinline__ void zero_acc(v8f (&acc)[2][2]) {
  #pragma unroll
  for (int i = 0; i < 2; ++i)
    #pragma unroll
    for (int j = 0; j < 2; ++j)
      #pragma unroll
      for (int e = 0; e < 8; ++e) acc[i][j][e] = 0.0f;
}

// ---------------------------------------------------------------------------
// feat = img(bf16)[256,2048] @ W1(bf16)[1024,2048]^T + b1  -> bf16 [256,1024]
// ---------------------------------------------------------------------------
__global__ __launch_bounds__(256)
void k_feat_gemm(const bf16* __restrict__ img, const bf16* __restrict__ W1bf,
                 const float* __restrict__ b1, bf16* __restrict__ feat)
{
  __shared__ __align__(16) bf16 As[2 * ASTRIDE];
  __shared__ __align__(16) bf16 Bs[2 * BSTRIDE];
  v8f acc[2][2]; zero_acc(acc);
  const int rowBase = blockIdx.y * BM;
  const int colBase = blockIdx.x * BN;
  gemm_k_loop<false>(img, W1bf, FT_SZ, FT_SZ, FT_SZ, 0, rowBase, colBase, As, Bs, acc);

  const int lane = threadIdx.x & 31, wave = threadIdx.x >> 5;
  const int wm = wave >> 1, wn = wave & 1, half = lane >> 4, l16 = lane & 15;
  #pragma unroll
  for (int i = 0; i < 2; ++i)
    #pragma unroll
    for (int j = 0; j < 2; ++j) {
      int col = colBase + wn * 32 + j * 16 + l16;
      #pragma unroll
      for (int e = 0; e < 8; ++e) {
        int row = rowBase + wm * 32 + i * 16 + half * 8 + e;
        feat[(size_t)row * EMB_SZ + col] = (bf16)(acc[i][j][e] + b1[col]);
      }
    }
}

// ---------------------------------------------------------------------------
// gates = X @ W_ih^T + H @ W_hh^T + b_ih + b_hh  -> fp32 [256, 4096]
// ---------------------------------------------------------------------------
__global__ __launch_bounds__(256)
void k_gates(const bf16* __restrict__ X,    const bf16* __restrict__ Wih,
             const bf16* __restrict__ Hprev, const bf16* __restrict__ Whh,
             const float* __restrict__ bih, const float* __restrict__ bhh,
             float* __restrict__ gates)
{
  __shared__ __align__(16) bf16 As[2 * ASTRIDE];
  __shared__ __align__(16) bf16 Bs[2 * BSTRIDE];
  v8f acc[2][2]; zero_acc(acc);
  const int rowBase = blockIdx.y * BM;
  const int colBase = blockIdx.x * BN;
  gemm_k_loop<false>(X,     Wih, EMB_SZ, EMB_SZ, EMB_SZ, 0, rowBase, colBase, As, Bs, acc);
  gemm_k_loop<false>(Hprev, Whh, MEM_SZ, MEM_SZ, MEM_SZ, 0, rowBase, colBase, As, Bs, acc);

  const int lane = threadIdx.x & 31, wave = threadIdx.x >> 5;
  const int wm = wave >> 1, wn = wave & 1, half = lane >> 4, l16 = lane & 15;
  #pragma unroll
  for (int i = 0; i < 2; ++i)
    #pragma unroll
    for (int j = 0; j < 2; ++j) {
      int col = colBase + wn * 32 + j * 16 + l16;
      float bias = bih[col] + bhh[col];
      #pragma unroll
      for (int e = 0; e < 8; ++e) {
        int row = rowBase + wm * 32 + i * 16 + half * 8 + e;
        gates[(size_t)row * (4 * MEM_SZ) + col] = acc[i][j][e] + bias;
      }
    }
}

// ---------------------------------------------------------------------------
// Pointwise LSTM cell update. gates fp32 [256,4096] (i,f,g,o), c fp32 [256,1024]
// h out -> bf16 [256,1024] (slot t of the H-stack)
// ---------------------------------------------------------------------------
__device__ __forceinline__ float sigmoidf_(float x) { return 1.0f / (1.0f + __expf(-x)); }

__global__ __launch_bounds__(256)
void k_lstm_pw(const float* __restrict__ gates, float* __restrict__ c,
               bf16* __restrict__ hout)
{
  int idx = blockIdx.x * blockDim.x + threadIdx.x;   // 0 .. 256*1024-1
  int b = idx >> 10, n = idx & 1023;
  const float* g = gates + (size_t)b * (4 * MEM_SZ);
  float ig = sigmoidf_(g[n]);
  float fg = sigmoidf_(g[MEM_SZ + n]);
  float gg = tanhf(g[2 * MEM_SZ + n]);
  float og = sigmoidf_(g[3 * MEM_SZ + n]);
  float cn = fg * c[idx] + ig * gg;
  c[idx] = cn;
  hout[idx] = (bf16)(og * tanhf(cn));
}

// ---------------------------------------------------------------------------
// Vocab projection over ALL steps at once:
//   Hall [20*256, 1024] bf16 @ W2[10000,1024]^T + b2 -> out[b, t, v] fp32
// ---------------------------------------------------------------------------
__global__ __launch_bounds__(256)
void k_vocab(const bf16* __restrict__ Hall, const bf16* __restrict__ W2bf,
             const float* __restrict__ b2, float* __restrict__ out)
{
  __shared__ __align__(16) bf16 As[2 * ASTRIDE];
  __shared__ __align__(16) bf16 Bs[2 * BSTRIDE];
  v8f acc[2][2]; zero_acc(acc);
  const int rowBase = blockIdx.y * BM;
  const int colBase = blockIdx.x * BN;
  gemm_k_loop<true>(Hall, W2bf, MEM_SZ, MEM_SZ, MEM_SZ, VOC_SZ, rowBase, colBase, As, Bs, acc);

  const int lane = threadIdx.x & 31, wave = threadIdx.x >> 5;
  const int wm = wave >> 1, wn = wave & 1, half = lane >> 4, l16 = lane & 15;
  #pragma unroll
  for (int i = 0; i < 2; ++i)
    #pragma unroll
    for (int j = 0; j < 2; ++j) {
      int col = colBase + wn * 32 + j * 16 + l16;
      if (col >= VOC_SZ) continue;
      float bias = b2[col];
      #pragma unroll
      for (int e = 0; e < 8; ++e) {
        int m = rowBase + wm * 32 + i * 16 + half * 8 + e;  // m = t*256 + b
        int t = m >> 8, b = m & 255;
        out[(size_t)b * (T_SZ * VOC_SZ) + (size_t)t * VOC_SZ + col] = acc[i][j][e] + bias;
      }
    }
}

// ---------------------------------------------------------------------------
// Small helpers: fp32->bf16 convert, zero-fill, embedding gather, lengths clip
// ---------------------------------------------------------------------------
__global__ void k_cvt(const float* __restrict__ s, bf16* __restrict__ d, int n)
{
  int i = blockIdx.x * blockDim.x + threadIdx.x;
  if (i < n) d[i] = (bf16)s[i];
}

__global__ void k_zero32(unsigned* __restrict__ p, int n)
{
  int i = blockIdx.x * blockDim.x + threadIdx.x;
  if (i < n) p[i] = 0u;
}

// Xemb[t][b][:] = bf16(emb_table[captions[b][t]])  for t in [0, T-1)
__global__ void k_embed(const float* __restrict__ emb, const int* __restrict__ caps,
                        bf16* __restrict__ X)
{
  int t = blockIdx.x / B_SZ;
  int b = blockIdx.x % B_SZ;
  int tok = caps[b * T_SZ + t];
  const float* src = emb + (size_t)tok * EMB_SZ;
  bf16* dst = X + ((size_t)t * B_SZ + b) * EMB_SZ;
  for (int e = threadIdx.x; e < EMB_SZ; e += blockDim.x) dst[e] = (bf16)src[e];
}

__global__ void k_lengths(const int* __restrict__ lens, int* __restrict__ out)
{
  int i = blockIdx.x * blockDim.x + threadIdx.x;
  if (i < B_SZ) { int v = lens[i]; out[i] = v < T_SZ ? v : T_SZ; }
}

// ---------------------------------------------------------------------------
extern "C" void kernel_launch(void* const* d_in, const int* in_sizes, int n_in,
                              void* d_out, int out_size, void* d_ws, size_t ws_size,
                              hipStream_t stream)
{
  const float* img  = (const float*)d_in[0];
  const int*   caps = (const int*)  d_in[1];
  const int*   lens = (const int*)  d_in[2];
  const float* W1   = (const float*)d_in[3];
  const float* b1   = (const float*)d_in[4];
  const float* emb  = (const float*)d_in[5];
  const float* Wih  = (const float*)d_in[6];
  const float* Whh  = (const float*)d_in[7];
  const float* bih  = (const float*)d_in[8];
  const float* bhh  = (const float*)d_in[9];
  const float* W2   = (const float*)d_in[10];
  const float* b2   = (const float*)d_in[11];
  float* out = (float*)d_out;

  char* ws = (char*)d_ws;
  size_t off = 0;
  auto take = [&](size_t bytes) -> char* {
    char* p = ws + off;
    off += (bytes + 255) & ~(size_t)255;
    return p;
  };
  bf16* W1bf   = (bf16*)take((size_t)EMB_SZ * FT_SZ * 2);
  bf16* Wihbf  = (bf16*)take((size_t)4 * MEM_SZ * EMB_SZ * 2);
  bf16* Whhbf  = (bf16*)take((size_t)4 * MEM_SZ * MEM_SZ * 2);
  bf16* W2bf   = (bf16*)take((size_t)VOC_SZ * MEM_SZ * 2);
  bf16* imgbf  = (bf16*)take((size_t)B_SZ * FT_SZ * 2);
  bf16* featbf = (bf16*)take((size_t)B_SZ * EMB_SZ * 2);
  bf16* Xemb   = (bf16*)take((size_t)(T_SZ - 1) * B_SZ * EMB_SZ * 2);
  bf16* hzero  = (bf16*)take((size_t)B_SZ * MEM_SZ * 2);
  bf16* Hall   = (bf16*)take((size_t)T_SZ * B_SZ * MEM_SZ * 2);
  float* cstat = (float*)take((size_t)B_SZ * MEM_SZ * 4);
  float* gates = (float*)take((size_t)B_SZ * 4 * MEM_SZ * 4);

  // --- weight / activation conversion (fp32 -> bf16), every call ---
  {
    int n;
    n = EMB_SZ * FT_SZ;        k_cvt<<<(n + 255) / 256, 256, 0, stream>>>(W1,  W1bf,  n);
    n = 4 * MEM_SZ * EMB_SZ;   k_cvt<<<(n + 255) / 256, 256, 0, stream>>>(Wih, Wihbf, n);
    n = 4 * MEM_SZ * MEM_SZ;   k_cvt<<<(n + 255) / 256, 256, 0, stream>>>(Whh, Whhbf, n);
    n = VOC_SZ * MEM_SZ;       k_cvt<<<(n + 255) / 256, 256, 0, stream>>>(W2,  W2bf,  n);
    n = B_SZ * FT_SZ;          k_cvt<<<(n + 255) / 256, 256, 0, stream>>>(img, imgbf, n);
  }
  // --- zero initial h (bf16 zeros == u32 zeros) and c ---
  {
    int nh = (B_SZ * MEM_SZ * 2) / 4;
    k_zero32<<<(nh + 255) / 256, 256, 0, stream>>>((unsigned*)hzero, nh);
    int nc = B_SZ * MEM_SZ;
    k_zero32<<<(nc + 255) / 256, 256, 0, stream>>>((unsigned*)cstat, nc);
  }
  // --- gather caption embeddings for steps 1..19 ---
  k_embed<<<(T_SZ - 1) * B_SZ, 256, 0, stream>>>(emb, caps, Xemb);

  // --- feat = img @ W1^T + b1 (bf16 out) ---
  k_feat_gemm<<<dim3(EMB_SZ / BN, B_SZ / BM), 256, 0, stream>>>(imgbf, W1bf, b1, featbf);

  // --- 20 sequential LSTM steps ---
  const size_t BH = (size_t)B_SZ * MEM_SZ;
  for (int t = 0; t < T_SZ; ++t) {
    const bf16* X  = (t == 0) ? featbf : (Xemb + (size_t)(t - 1) * BH);
    const bf16* Hp = (t == 0) ? hzero  : (Hall + (size_t)(t - 1) * BH);
    k_gates<<<dim3(4 * MEM_SZ / BN, B_SZ / BM), 256, 0, stream>>>(
        X, Wihbf, Hp, Whhbf, bih, bhh, gates);
    k_lstm_pw<<<(B_SZ * MEM_SZ) / 256, 256, 0, stream>>>(
        gates, cstat, Hall + (size_t)t * BH);
  }

  // --- batched vocab projection for all 20 steps, remapped to [b, t, v] ---
  k_vocab<<<dim3((VOC_SZ + BN - 1) / BN, (T_SZ * B_SZ) / BM), 256, 0, stream>>>(
      Hall, W2bf, b2, out);

  // --- clipped lengths appended after the logits ---
  int* lenOut = (int*)d_out + (size_t)B_SZ * T_SZ * VOC_SZ;
  k_lengths<<<1, 256, 0, stream>>>(lens, lenOut);
}
